// SelfAttentionHead_67929202753637
// MI455X (gfx1250) — matrix operand order repaired and checked by reading, not compiled, not run on hardware
//
#include <hip/hip_runtime.h>

// ---------------- CDNA5 WMMA types ----------------
typedef __attribute__((ext_vector_type(16))) __bf16 v16bf;
typedef __attribute__((ext_vector_type(8)))  __bf16 v8bf;
typedef __attribute__((ext_vector_type(8)))  float  v8f;

#define T_SEQ 256
#define C_DIM 64
#define H_DIM 64
#define NEGBIG (-1.0e30f)
// 1/sqrt(H) * log2(e): Q pre-scaled so S is in log2 domain
#define QSCALE 0.18033688011112042f

__device__ __forceinline__ v8f wmma_bf16(v16bf a, v16bf b, v8f c) {
    // D = A(16x32) * B(32x16) + C, f32 accumulate
    return __builtin_amdgcn_wmma_f32_16x16x32_bf16(false, a, false, b,
                                                   (short)0, c, false, false);
}

// A fragment (16x32 bf16) from a row-major bf16 LDS matrix.
// lane(g = lane>>4, m = lane&15): elems 0..7 -> k = g*8+e ; elems 8..15 -> k = 16+g*8+e
__device__ __forceinline__ v16bf load_a_lds(const __bf16* base, int row0, int k0,
                                            int stride, int lane) {
    int m = lane & 15, g = lane >> 4;
    const __bf16* src = base + (row0 + m) * stride + k0 + g * 8;
    v8bf lo = *(const v8bf*)(src);
    v8bf hi = *(const v8bf*)(src + 16);
    v16bf a;
#pragma unroll
    for (int e = 0; e < 8; ++e) { a[e] = lo[e]; a[8 + e] = hi[e]; }
    return a;
}

// A fragment (16x32) from a row-major fp32 GLOBAL matrix (stride C_DIM), cvt->bf16
__device__ __forceinline__ v16bf load_a_gf32(const float* base, int row0, int k0, int lane) {
    int m = lane & 15, g = lane >> 4;
    const float* src = base + (row0 + m) * C_DIM + k0 + g * 8;
    v16bf a;
#pragma unroll
    for (int e = 0; e < 8; ++e) {
        a[e]     = (__bf16)src[e];
        a[8 + e] = (__bf16)src[16 + e];
    }
    return a;
}

// B fragment where B[k][n] = S[row0+n][k0+k], S row-major bf16 in LDS.
// lane(g,n): elems 0..7 -> k = g*8+e ; elems 8..15 -> k = 16+g*8+e
__device__ __forceinline__ v16bf load_bt_lds(const __bf16* base, int row0, int k0,
                                             int stride, int lane) {
    int n = lane & 15, g = lane >> 4;
    const __bf16* src = base + (row0 + n) * stride + k0 + g * 8;
    v8bf lo = *(const v8bf*)(src);
    v8bf hi = *(const v8bf*)(src + 16);
    v16bf b;
#pragma unroll
    for (int e = 0; e < 8; ++e) { b[e] = lo[e]; b[8 + e] = hi[e]; }
    return b;
}

// Store C-layout v8f accumulator (16x16) to row-major bf16 LDS
__device__ __forceinline__ void store_c_bf16(v8f acc, __bf16* base, int row0, int col0,
                                             int stride, int lane) {
    int n = lane & 15, g = lane >> 4;
#pragma unroll
    for (int r = 0; r < 8; ++r)
        base[(row0 + r + 8 * g) * stride + col0 + n] = (__bf16)acc[r];
}

// Store C-layout accumulator TRANSPOSED: dst[col][row]
__device__ __forceinline__ void store_c_bf16_T(v8f acc, __bf16* base, int row0, int col0,
                                               int stride, int lane) {
    int n = lane & 15, g = lane >> 4;
#pragma unroll
    for (int r = 0; r < 8; ++r)
        base[(col0 + n) * stride + row0 + r + 8 * g] = (__bf16)acc[r];
}

__global__ __launch_bounds__(256)
void attn_head_fused_kernel(const float* __restrict__ x,
                            const float* __restrict__ Wq,
                            const float* __restrict__ Wk,
                            const float* __restrict__ Wv,
                            float* __restrict__ out) {
    __shared__ __align__(64) __bf16 QB[T_SEQ * H_DIM];        // 32 KB (Q pre-scaled)
    __shared__ __align__(64) __bf16 KB[T_SEQ * H_DIM];        // 32 KB
    __shared__ __align__(64) __bf16 VTB[H_DIM * T_SEQ];       // 32 KB, V transposed [h][t]
    __shared__ __align__(64) __bf16 WBUF[3 * H_DIM * C_DIM];  // 24 KB; phase2: P scratch

    const int bidx = blockIdx.x;
    const int tid  = threadIdx.x;
    const int lane = tid & 31;
    const int wave = __builtin_amdgcn_readfirstlane(tid >> 5);  // provably uniform
    const float* xb = x + (size_t)bidx * T_SEQ * C_DIM;

    // ---- stage weights (bf16) into LDS ----
    for (int idx = tid; idx < H_DIM * C_DIM; idx += 256) {
        WBUF[idx]                     = (__bf16)Wq[idx];
        WBUF[H_DIM * C_DIM + idx]     = (__bf16)Wk[idx];
        WBUF[2 * H_DIM * C_DIM + idx] = (__bf16)Wv[idx];
    }
    __syncthreads();

    // ---- phase 1: Q = xWq^T * QSCALE, K = xWk^T, V^T = (xWv^T)^T ----
    for (int ph = 0; ph < 2; ++ph) {
        int i    = (ph == 0) ? wave : 15 - wave;   // 16 row tiles over 8 waves
        int row0 = i * 16;
        v16bf ax0 = load_a_gf32(xb, row0, 0, lane);
        v16bf ax1 = load_a_gf32(xb, row0, 32, lane);
#pragma unroll
        for (int m = 0; m < 3; ++m) {
            const __bf16* w = WBUF + m * H_DIM * C_DIM;
#pragma unroll
            for (int c = 0; c < 4; ++c) {
                v8f acc = {};
                acc = wmma_bf16(ax0, load_bt_lds(w, c * 16, 0, C_DIM, lane), acc);
                acc = wmma_bf16(ax1, load_bt_lds(w, c * 16, 32, C_DIM, lane), acc);
                if (m == 0) {
#pragma unroll
                    for (int r = 0; r < 8; ++r) acc[r] *= QSCALE;
                    store_c_bf16(acc, QB, row0, c * 16, H_DIM, lane);
                } else if (m == 1) {
                    store_c_bf16(acc, KB, row0, c * 16, H_DIM, lane);
                } else {
                    store_c_bf16_T(acc, VTB, row0, c * 16, T_SEQ, lane);
                }
            }
        }
    }
    __syncthreads();  // also retires WBUF reads before it becomes P scratch

    // ---- phase 2: causal flash attention, 64 key-cols per step ----
    __bf16* psc = WBUF + wave * (16 * 64);   // per-wave 16x64 bf16 P scratch (16 KB total)
    const int n = lane & 15, g = lane >> 4;

    v16bf ones;
#pragma unroll
    for (int e = 0; e < 16; ++e) ones[e] = (__bf16)1.0f;

    for (int ph = 0; ph < 2; ++ph) {
        int i    = (ph == 0) ? wave : 15 - wave;
        int row0 = i * 16;
        v16bf aq0 = load_a_lds(QB, row0, 0, H_DIM, lane);
        v16bf aq1 = load_a_lds(QB, row0, 32, H_DIM, lane);
        v8f o0 = {}, o1 = {}, o2 = {}, o3 = {};
        v8f lsum = {};   // row-sum accumulator via ones-WMMA (flash "l")
        v8f mrow;
#pragma unroll
        for (int r = 0; r < 8; ++r) mrow[r] = NEGBIG;

        int jmax = row0 >> 6;
        for (int j = 0; j <= jmax; ++j) {
            int col0 = j * 64;
            // ---- S block: 16 rows x 64 cols (4 sub-tiles), log2 domain ----
            v8f s[4];
#pragma unroll
            for (int t = 0; t < 4; ++t) {
                int cb = col0 + 16 * t;              // uniform per wave
                if (cb <= row0 + 15) {               // sub-tile intersects causal region
                    v8f st = {};
                    st = wmma_bf16(aq0, load_bt_lds(KB, cb, 0, H_DIM, lane), st);
                    st = wmma_bf16(aq1, load_bt_lds(KB, cb, 32, H_DIM, lane), st);
                    if (cb + 15 > row0) {            // diagonal sub-tile: element mask
#pragma unroll
                        for (int r = 0; r < 8; ++r)
                            if (cb + n > row0 + r + 8 * g) st[r] = NEGBIG;
                    }
                    s[t] = st;
                } else {                             // fully masked: skip the WMMAs
#pragma unroll
                    for (int r = 0; r < 8; ++r) s[t][r] = NEGBIG;
                }
            }
            // ---- online softmax: row max over 64 cols ----
            v8f newm;
#pragma unroll
            for (int r = 0; r < 8; ++r)
                newm[r] = fmaxf(fmaxf(s[0][r], s[1][r]), fmaxf(s[2][r], s[3][r]));
#pragma unroll
            for (int d = 1; d < 16; d <<= 1) {
#pragma unroll
                for (int r = 0; r < 8; ++r)
                    newm[r] = fmaxf(newm[r], __shfl_xor(newm[r], d, 32));
            }
            v8f f;
#pragma unroll
            for (int r = 0; r < 8; ++r) {
                newm[r] = fmaxf(newm[r], mrow[r]);
                f[r]    = __builtin_amdgcn_exp2f(mrow[r] - newm[r]);
                mrow[r] = newm[r];
            }
#pragma unroll
            for (int t = 0; t < 4; ++t)
#pragma unroll
                for (int r = 0; r < 8; ++r)
                    s[t][r] = __builtin_amdgcn_exp2f(s[t][r] - newm[r]);
#pragma unroll
            for (int r = 0; r < 8; ++r) {
                o0[r] *= f[r]; o1[r] *= f[r]; o2[r] *= f[r]; o3[r] *= f[r];
                lsum[r] *= f[r];
            }
            // ---- P (C layout) -> bf16 LDS scratch -> A fragments ----
#pragma unroll
            for (int t = 0; t < 4; ++t)
#pragma unroll
                for (int r = 0; r < 8; ++r)
                    psc[(r + 8 * g) * 64 + 16 * t + n] = (__bf16)s[t][r];
            v16bf ap0 = load_a_lds(psc, 0, 0, 64, lane);
            v16bf ap1 = load_a_lds(psc, 0, 32, 64, lane);
            // ---- O += P @ V ; l += P @ 1 (row sums by WMMA) ----
            o0 = wmma_bf16(ap0, load_bt_lds(VTB, 0,  col0, T_SEQ, lane), o0);
            o1 = wmma_bf16(ap0, load_bt_lds(VTB, 16, col0, T_SEQ, lane), o1);
            o2 = wmma_bf16(ap0, load_bt_lds(VTB, 32, col0, T_SEQ, lane), o2);
            o3 = wmma_bf16(ap0, load_bt_lds(VTB, 48, col0, T_SEQ, lane), o3);
            lsum = wmma_bf16(ap0, ones, lsum);
            o0 = wmma_bf16(ap1, load_bt_lds(VTB, 0,  col0 + 32, T_SEQ, lane), o0);
            o1 = wmma_bf16(ap1, load_bt_lds(VTB, 16, col0 + 32, T_SEQ, lane), o1);
            o2 = wmma_bf16(ap1, load_bt_lds(VTB, 32, col0 + 32, T_SEQ, lane), o2);
            o3 = wmma_bf16(ap1, load_bt_lds(VTB, 48, col0 + 32, T_SEQ, lane), o3);
            lsum = wmma_bf16(ap1, ones, lsum);
        }
        // ---- epilogue: divide by row sum, write fp32 out ----
        float* ob = out + ((size_t)bidx * T_SEQ + row0) * H_DIM;
#pragma unroll
        for (int r = 0; r < 8; ++r) {
            float inv = __builtin_amdgcn_rcpf(lsum[r]);
            int row = r + 8 * g;
            ob[row * H_DIM + 0  + n] = o0[r] * inv;
            ob[row * H_DIM + 16 + n] = o1[r] * inv;
            ob[row * H_DIM + 32 + n] = o2[r] * inv;
            ob[row * H_DIM + 48 + n] = o3[r] * inv;
        }
    }
}

extern "C" void kernel_launch(void* const* d_in, const int* in_sizes, int n_in,
                              void* d_out, int out_size, void* d_ws, size_t ws_size,
                              hipStream_t stream) {
    const float* x  = (const float*)d_in[0];
    const float* Wq = (const float*)d_in[1];
    const float* Wk = (const float*)d_in[2];
    const float* Wv = (const float*)d_in[3];
    float* out = (float*)d_out;
    int B = in_sizes[0] / (T_SEQ * C_DIM);  // 2048
    hipLaunchKernelGGL(attn_head_fused_kernel, dim3(B), dim3(256), 0, stream,
                       x, Wq, Wk, Wv, out);
}
